// TableGNN_55843164782682
// MI455X (gfx1250) — matrix operand order repaired and checked by reading, not compiled, not run on hardware
//
#include <hip/hip_runtime.h>
#include <hip/hip_bf16.h>

typedef _Float16 v16h __attribute__((ext_vector_type(16)));
typedef _Float16 v8h  __attribute__((ext_vector_type(8)));
typedef float    v8f  __attribute__((ext_vector_type(8)));

// ---------------- WMMA helpers (CDNA5, wave32) ----------------

__device__ __forceinline__ v8f wmma_f16(v16h a, v16h b, v8f c) {
    // D = A(16x32 f16) * B(32x16 f16) + C(16x16 f32)
    return __builtin_amdgcn_wmma_f32_16x16x32_f16(
        /*neg_a=*/false, a, /*neg_b=*/false, b,
        /*c_mod=*/(short)0, c, /*reuse_a=*/false, /*reuse_b=*/false);
}

// A fragment from an LDS tile [16][64] of f16, K-window starting at k0 (0 or 32).
__device__ __forceinline__ v16h ldsA(const _Float16 (*tile)[64], int lane, int k0) {
    int row = lane & 15;
    int hi  = (lane >> 4) & 1;
    const _Float16* p = &tile[row][k0 + hi * 8];
    v8h lo  = *(const v8h*)p;
    v8h hi8 = *(const v8h*)(p + 16);
    v16h a;
#pragma unroll
    for (int i = 0; i < 8; ++i) { a[i] = lo[i]; a[8 + i] = hi8[i]; }
    return a;
}

// B fragment from fp16 transposed weights wt[out][64] (K contiguous).
__device__ __forceinline__ v16h gblB(const _Float16* wt, int lane, int nbase, int k0) {
    int col = nbase + (lane & 15);
    int hi  = (lane >> 4) & 1;
    const _Float16* p = wt + (size_t)col * 64 + k0 + hi * 8;
    v8h lo  = *(const v8h*)p;
    v8h hi8 = *(const v8h*)(p + 16);
    v16h b;
#pragma unroll
    for (int i = 0; i < 8; ++i) { b[i] = lo[i]; b[8 + i] = hi8[i]; }
    return b;
}

// 64->64 layer on one 16-node tile: out = relu(in @ W + b), in/out f16 LDS tiles.
__device__ __forceinline__ void layer64(const _Float16 (*in)[64], _Float16 (*out)[64],
                                        const _Float16* wt, const float* bias,
                                        int lane) {
    int hi = lane >> 4, hl = lane & 15;
    v16h a0 = ldsA(in, lane, 0);     // hoisted: reused across all 4 output tiles
    v16h a1 = ldsA(in, lane, 32);
#pragma unroll
    for (int nb = 0; nb < 4; ++nb) {
        v8f c = {};
        c = wmma_f16(a0, gblB(wt, lane, nb * 16, 0),  c);
        c = wmma_f16(a1, gblB(wt, lane, nb * 16, 32), c);
        int col = nb * 16 + hl;
        float bv = bias[col];
#pragma unroll
        for (int r = 0; r < 8; ++r) {
            float v = c[r] + bv;
            v = v > 0.0f ? v : 0.0f;
            out[r + hi * 8][col] = (_Float16)v;  // D: VGPR r -> row r+8*hi, col=lane%16
        }
    }
    asm volatile("s_wait_dscnt 0" ::: "memory");
}

// 64->32 head layer: out = relu(in @ W + b) into f32 LDS tile [16][32].
__device__ __forceinline__ void layer32(const _Float16 (*in)[64], float (*out)[32],
                                        const _Float16* wt, const float* bias, int lane) {
    int hi = lane >> 4, hl = lane & 15;
    v16h a0 = ldsA(in, lane, 0);
    v16h a1 = ldsA(in, lane, 32);
#pragma unroll
    for (int nb = 0; nb < 2; ++nb) {
        v8f c = {};
        c = wmma_f16(a0, gblB(wt, lane, nb * 16, 0),  c);
        c = wmma_f16(a1, gblB(wt, lane, nb * 16, 32), c);
        int col = nb * 16 + hl;
        float bv = bias[col];
#pragma unroll
        for (int r = 0; r < 8; ++r) {
            float v = c[r] + bv;
            out[r + hi * 8][col] = v > 0.0f ? v : 0.0f;
        }
    }
    asm volatile("s_wait_dscnt 0" ::: "memory");
}

// ---------------- kernels ----------------

// Convert/transpose the big weight matrices to fp16 [out][64] layout.
__global__ void gnn_prep(const float* W2, const float* Wf1, const float* Wf2,
                         const float* Ws1, const float* Wt1,
                         _Float16* wt2, _Float16* wtf1, _Float16* wtf2,
                         _Float16* wts1, _Float16* wtt1) {
    int i = blockIdx.x * blockDim.x + threadIdx.x;
    if (i < 4096)       { int o = i >> 6, k = i & 63;                   wt2 [o * 64 + k] = (_Float16)W2 [k * 64 + o]; }
    else if (i < 8192)  { int j = i - 4096;  int o = j >> 6, k = j & 63; wtf1[o * 64 + k] = (_Float16)Wf1[k * 64 + o]; }
    else if (i < 12288) { int j = i - 8192;  int o = j >> 6, k = j & 63; wtf2[o * 64 + k] = (_Float16)Wf2[k * 64 + o]; }
    else if (i < 14336) { int j = i - 12288; int o = j >> 6, k = j & 63; wts1[o * 64 + k] = (_Float16)Ws1[k * 32 + o]; }
    else if (i < 16384) { int j = i - 14336; int o = j >> 6, k = j & 63; wtt1[o * 64 + k] = (_Float16)Wt1[k * 32 + o]; }
}

__global__ void gnn_zero(float* p, size_t n) {
    size_t i = blockIdx.x * (size_t)blockDim.x + threadIdx.x;
    size_t stride = (size_t)gridDim.x * blockDim.x;
    for (; i < n; i += stride) p[i] = 0.0f;
}

// Cell encoder: h = relu(x@W1+b1)@W2+b2, one 16-node tile per wave.
__global__ void gnn_encoder(const float* __restrict__ x, const float* __restrict__ W1,
                            const float* __restrict__ b1, const _Float16* __restrict__ wt2,
                            const float* __restrict__ b2, float* __restrict__ h,
                            int nTiles, int N) {
    __shared__ _Float16 t1[8][16][64];
    int lane = threadIdx.x & 31;
    int w    = threadIdx.x >> 5;
    int wave = (blockIdx.x * blockDim.x + threadIdx.x) >> 5;
    int wavesTotal = (gridDim.x * blockDim.x) >> 5;

    for (int tile = wave; tile < nTiles; tile += wavesTotal) {
        int  base = tile * 16;
        bool full = (base + 16 <= N);            // wave-uniform fast path
        // layer1 (K=6, scalar): lane<16 -> row=lane cols 0..31; lane>=16 -> row cols 32..63
        int row   = lane & 15;
        int cbase = (lane >> 4) * 32;
        int node  = base + row; if (node >= N) node = N - 1;
        float xr[6];
#pragma unroll
        for (int k = 0; k < 6; ++k) xr[k] = x[(size_t)node * 6 + k];
#pragma unroll 4
        for (int c = 0; c < 32; ++c) {
            int col = cbase + c;
            float acc = b1[col];
#pragma unroll
            for (int k = 0; k < 6; ++k) acc += xr[k] * W1[k * 64 + col];
            t1[w][row][col] = (_Float16)(acc > 0.0f ? acc : 0.0f);
        }
        asm volatile("s_wait_dscnt 0" ::: "memory");
        // layer2 (64x64) via WMMA, write h fp32
        int hi = lane >> 4, hl = lane & 15;
        v16h a0 = ldsA(t1[w], lane, 0);
        v16h a1 = ldsA(t1[w], lane, 32);
#pragma unroll
        for (int nb = 0; nb < 4; ++nb) {
            v8f c = {};
            c = wmma_f16(a0, gblB(wt2, lane, nb * 16, 0),  c);
            c = wmma_f16(a1, gblB(wt2, lane, nb * 16, 32), c);
            int col = nb * 16 + hl;
            float bv = b2[col];
            if (full) {
#pragma unroll
                for (int r = 0; r < 8; ++r)
                    h[(size_t)(base + r + hi * 8) * 64 + col] = c[r] + bv;
            } else {
#pragma unroll
                for (int r = 0; r < 8; ++r) {
                    int nrow = base + r + hi * 8;
                    if (nrow < N) h[(size_t)nrow * 64 + col] = c[r] + bv;
                }
            }
        }
    }
}

// Edge scatter: agg[src] += h[dst]; counts[src] += 1.  One wave per edge.
// Edge indices are wave-uniform: force them through SGPRs so the adj loads
// become s_load_b64 (SMEM) instead of 32 replicated VMEM loads per edge.
__global__ void gnn_aggregate(const long long* __restrict__ adj, const float* __restrict__ h,
                              float* __restrict__ agg, float* __restrict__ counts,
                              long long E) {
    int lane   = threadIdx.x & 31;
    int wslot  = __builtin_amdgcn_readfirstlane(threadIdx.x >> 5);   // SGPR wave slot
    long long wave  = (long long)blockIdx.x * (blockDim.x >> 5) + wslot;
    long long waves = (long long)gridDim.x * (blockDim.x >> 5);
    for (long long e = wave; e < E; e += waves) {
        long long s = adj[e];       // scalar load: src (segment index)
        long long d = adj[E + e];   // scalar load: dst (gathered)
        float v0 = h[d * 64 + lane];
        float v1 = h[d * 64 + 32 + lane];
        atomicAdd(&agg[s * 64 + lane], v0);
        atomicAdd(&agg[s * 64 + 32 + lane], v1);
        if (lane == 0) atomicAdd(&counts[s], 1.0f);
    }
}

// h' = h + agg/max(cnt,1); two 64->64 relu layers; two heads; write outputs.
__global__ void gnn_fused_head(const float* __restrict__ h, const float* __restrict__ agg,
                               const float* __restrict__ counts,
                               const _Float16* __restrict__ wtf1, const float* __restrict__ bf1,
                               const _Float16* __restrict__ wtf2, const float* __restrict__ bf2,
                               const _Float16* __restrict__ wts1, const float* __restrict__ bs1,
                               const float* __restrict__ Ws2, const float* __restrict__ bs2,
                               const _Float16* __restrict__ wtt1, const float* __restrict__ bt1,
                               const float* __restrict__ Wt2, const float* __restrict__ bt2,
                               float* __restrict__ outScores, float* __restrict__ outTypes,
                               int nTiles, int N) {
    __shared__ _Float16 tA[8][16][64];
    __shared__ _Float16 tB[8][16][64];
    __shared__ float    hs[8][16][32];
    __shared__ float    ht[8][16][32];
    int lane = threadIdx.x & 31;
    int w    = threadIdx.x >> 5;
    int wave = (blockIdx.x * blockDim.x + threadIdx.x) >> 5;
    int wavesTotal = (gridDim.x * blockDim.x) >> 5;

    for (int tile = wave; tile < nTiles; tile += wavesTotal) {
        int  base = tile * 16;
        bool full = (base + 16 <= N);
        int row   = lane & 15;
        int cbase = (lane >> 4) * 32;
        int node  = base + row; if (node >= N) node = N - 1;
        float inv = 1.0f / fmaxf(counts[node], 1.0f);
        const float4* hv = (const float4*)(h   + (size_t)node * 64 + cbase);
        const float4* av = (const float4*)(agg + (size_t)node * 64 + cbase);
#pragma unroll
        for (int c4 = 0; c4 < 8; ++c4) {
            float4 hh = hv[c4], aa = av[c4];
            int c = cbase + c4 * 4;
            tA[w][row][c + 0] = (_Float16)(hh.x + aa.x * inv);
            tA[w][row][c + 1] = (_Float16)(hh.y + aa.y * inv);
            tA[w][row][c + 2] = (_Float16)(hh.z + aa.z * inv);
            tA[w][row][c + 3] = (_Float16)(hh.w + aa.w * inv);
        }
        asm volatile("s_wait_dscnt 0" ::: "memory");

        layer64(tA[w], tB[w], wtf1, bf1, lane);   // h = relu(h@Wf1+bf1)
        layer64(tB[w], tA[w], wtf2, bf2, lane);   // h = relu(h@Wf2+bf2)
        layer32(tA[w], hs[w], wts1, bs1, lane);   // s1 = relu(h@Ws1+bs1)
        layer32(tA[w], ht[w], wtt1, bt1, lane);   // t1 = relu(h@Wt1+bt1)

        if (lane < 16) {
            int n2 = base + lane;
            if (full || n2 < N) {
                float sc = bs2[0];
#pragma unroll
                for (int k = 0; k < 32; ++k) sc += hs[w][lane][k] * Ws2[k];
                outScores[n2] = sc;
                float o0 = bt2[0], o1 = bt2[1], o2 = bt2[2], o3 = bt2[3];
#pragma unroll
                for (int k = 0; k < 32; ++k) {
                    float v = ht[w][lane][k];
                    o0 += v * Wt2[k * 4 + 0];
                    o1 += v * Wt2[k * 4 + 1];
                    o2 += v * Wt2[k * 4 + 2];
                    o3 += v * Wt2[k * 4 + 3];
                }
                *(float4*)(outTypes + (size_t)n2 * 4) = make_float4(o0, o1, o2, o3);
            }
        }
    }
}

// ---------------- host launcher ----------------

extern "C" void kernel_launch(void* const* d_in, const int* in_sizes, int n_in,
                              void* d_out, int out_size, void* d_ws, size_t ws_size,
                              hipStream_t stream) {
    const float*     x   = (const float*)d_in[0];
    const long long* adj = (const long long*)d_in[1];
    const float* W1  = (const float*)d_in[2];
    const float* b1  = (const float*)d_in[3];
    const float* W2  = (const float*)d_in[4];
    const float* b2  = (const float*)d_in[5];
    const float* Wf1 = (const float*)d_in[6];
    const float* bf1 = (const float*)d_in[7];
    const float* Wf2 = (const float*)d_in[8];
    const float* bf2 = (const float*)d_in[9];
    const float* Ws1 = (const float*)d_in[10];
    const float* bs1 = (const float*)d_in[11];
    const float* Ws2 = (const float*)d_in[12];
    const float* bs2 = (const float*)d_in[13];
    const float* Wt1 = (const float*)d_in[14];
    const float* bt1 = (const float*)d_in[15];
    const float* Wt2 = (const float*)d_in[16];
    const float* bt2 = (const float*)d_in[17];

    const int       N = in_sizes[0] / 6;
    const long long E = (long long)in_sizes[1] / 2;

    // workspace layout (N*516 bytes stays 32B-aligned for the f16 weight blocks)
    float* hbuf   = (float*)d_ws;                 // N*64 fp32
    float* agg    = hbuf + (size_t)N * 64;        // N*64 fp32
    float* counts = agg  + (size_t)N * 64;        // N fp32
    _Float16* wt2  = (_Float16*)(counts + N);     // 64*64
    _Float16* wtf1 = wt2  + 64 * 64;
    _Float16* wtf2 = wtf1 + 64 * 64;
    _Float16* wts1 = wtf2 + 64 * 64;              // 32*64
    _Float16* wtt1 = wts1 + 32 * 64;              // 32*64

    float* outScores = (float*)d_out;
    float* outTypes  = outScores + N;

    const int nTiles = (N + 15) / 16;
    const int tileBlocks = (nTiles + 7) / 8;      // 8 waves (= 8 tiles) per 256-thread block

    gnn_prep<<<64, 256, 0, stream>>>(W2, Wf1, Wf2, Ws1, Wt1, wt2, wtf1, wtf2, wts1, wtt1);
    gnn_zero<<<2048, 256, 0, stream>>>(agg, (size_t)N * 65);   // agg + counts contiguous
    gnn_encoder<<<tileBlocks, 256, 0, stream>>>(x, W1, b1, wt2, b2, hbuf, nTiles, N);
    gnn_aggregate<<<4096, 256, 0, stream>>>(adj, hbuf, agg, counts, E);
    gnn_fused_head<<<tileBlocks, 256, 0, stream>>>(hbuf, agg, counts,
                                                   wtf1, bf1, wtf2, bf2,
                                                   wts1, bs1, Ws2, bs2,
                                                   wtt1, bt1, Wt2, bt2,
                                                   outScores, outTypes, nTiles, N);
    (void)n_in; (void)out_size; (void)ws_size;
}
